// MultiheadAttention_72121090834528
// MI455X (gfx1250) — matrix operand order repaired and checked by reading, not compiled, not run on hardware
//
#include <hip/hip_runtime.h>

// ---------------------------------------------------------------------------
// MHA forward for MI455X (gfx1250, wave32, WMMA).
//   S=2048 seq, D=2048 model, H=16 heads, DK=128 head dim.
//   out = softmax_causal((XWq^T/sqrt) (XWk^T/sqrt)^T) (XWv^T) Wo^T
// Compute-bound (~103 GFLOP vs ~130MB traffic) -> run on v_wmma_f32_16x16x32_f16.
// ---------------------------------------------------------------------------

#define S_DIM 2048
#define D_DIM 2048
#define H_NUM 16
#define DK_DIM 128

typedef __attribute__((ext_vector_type(16))) _Float16 v16h;
typedef __attribute__((ext_vector_type(8)))  _Float16 v8h;
typedef __attribute__((ext_vector_type(4)))  _Float16 v4h;
typedef __attribute__((ext_vector_type(8)))  float    v8f;

union V16H { v16h v; v8h h[2]; };

__device__ __forceinline__ v8f wmma_f16(v16h a, v16h b, v8f c) {
    return __builtin_amdgcn_wmma_f32_16x16x32_f16(
        /*neg_a=*/false, a, /*neg_b=*/false, b,
        /*c_mod=*/(short)0, c, /*reuse_a=*/false, /*reuse_b=*/false);
}

// ---------------------------------------------------------------------------
// Tiled GEMM: Y[M,N] = (X[M,K] @ W[N,K]^T) * scale   (M=N=K=2048)
// Block: 256 threads = 8 waves; tile BM=128, BN=128, BK=32.
// Wave (wm in 0..3, wn in 0..1) owns a 32x64 region = 2x4 WMMA tiles.
// A tile + B tile staged in LDS as f16.
// ---------------------------------------------------------------------------
template <bool A_F16, bool OUT_F16>
__global__ __launch_bounds__(256)
void gemm_xwT(const void* __restrict__ Xv, const float* __restrict__ W,
              void* __restrict__ Yv, float scale) {
    __shared__ __align__(16) _Float16 ldsA[128][32];
    __shared__ __align__(16) _Float16 ldsB[128][32];

    const int bm = blockIdx.y * 128;
    const int bn = blockIdx.x * 128;
    const int tid  = threadIdx.x;
    const int lane = tid & 31;
    const int wave = tid >> 5;      // 0..7
    const int wm   = wave >> 1;     // 0..3 -> +32 rows
    const int wn   = wave & 1;      // 0..1 -> +64 cols
    const int l15  = lane & 15;
    const int half = lane >> 4;     // 0/1

    v8f acc[2][4];
#pragma unroll
    for (int i = 0; i < 2; ++i)
#pragma unroll
        for (int j = 0; j < 4; ++j) acc[i][j] = (v8f){};

    for (int k = 0; k < D_DIM; k += 32) {
        __syncthreads();
        // ---- load A tile (128x32) -> f16 LDS ----
        if constexpr (A_F16) {
            const _Float16* X = (const _Float16*)Xv;
#pragma unroll
            for (int i = 0; i < 2; ++i) {
                int lin = i * 256 + tid;            // 0..511 chunks of 8 halves
                int row = lin >> 2;
                int c8  = (lin & 3) * 8;
                *(v8h*)&ldsA[row][c8] =
                    *(const v8h*)&X[(size_t)(bm + row) * D_DIM + k + c8];
            }
        } else {
            const float* X = (const float*)Xv;
#pragma unroll
            for (int i = 0; i < 4; ++i) {
                int lin = i * 256 + tid;            // 0..1023 chunks of 4 floats
                int row = lin >> 3;
                int c4  = (lin & 7) * 4;
                const float4 f = *(const float4*)&X[(size_t)(bm + row) * D_DIM + k + c4];
                v4h hv;
                hv[0] = (_Float16)f.x; hv[1] = (_Float16)f.y;
                hv[2] = (_Float16)f.z; hv[3] = (_Float16)f.w;
                *(v4h*)&ldsA[row][c4] = hv;
            }
        }
        // ---- load B tile: rows n of W (W is [N,K] row-major, k contiguous) ----
#pragma unroll
        for (int i = 0; i < 4; ++i) {
            int lin = i * 256 + tid;
            int row = lin >> 3;
            int c4  = (lin & 7) * 4;
            const float4 f = *(const float4*)&W[(size_t)(bn + row) * D_DIM + k + c4];
            v4h hv;
            hv[0] = (_Float16)f.x; hv[1] = (_Float16)f.y;
            hv[2] = (_Float16)f.z; hv[3] = (_Float16)f.w;
            *(v4h*)&ldsB[row][c4] = hv;
        }
        __syncthreads();

        // ---- fragments: same contiguous k order for A and B ----
        // A 16x32: row = lane&15 (both halves); K = half*16 .. +15 contiguous
        V16H af[2];
#pragma unroll
        for (int i = 0; i < 2; ++i) {
            const _Float16* p = &ldsA[wm * 32 + i * 16 + l15][half * 16];
            af[i].h[0] = *(const v8h*)p;
            af[i].h[1] = *(const v8h*)(p + 8);
        }
        // B 32x16: n = lane&15, K = half*16 .. +15 contiguous
        v16h bf[4];
#pragma unroll
        for (int j = 0; j < 4; ++j)
            bf[j] = *(const v16h*)&ldsB[wn * 64 + j * 16 + l15][half * 16];

#pragma unroll
        for (int i = 0; i < 2; ++i)
#pragma unroll
            for (int j = 0; j < 4; ++j)
                acc[i][j] = wmma_f16(af[i].v, bf[j], acc[i][j]);
    }

    // ---- store: C/D layout row = half*8 + vgpr, col = lane&15 ----
#pragma unroll
    for (int i = 0; i < 2; ++i)
#pragma unroll
        for (int j = 0; j < 4; ++j) {
            const int row0 = bm + wm * 32 + i * 16 + half * 8;
            const int col  = bn + wn * 64 + j * 16 + l15;
#pragma unroll
            for (int v = 0; v < 8; ++v) {
                float y = acc[i][j][v] * scale;
                if constexpr (OUT_F16)
                    ((_Float16*)Yv)[(size_t)(row0 + v) * D_DIM + col] = (_Float16)y;
                else
                    ((float*)Yv)[(size_t)(row0 + v) * D_DIM + col] = y;
            }
        }
}

// ---------------------------------------------------------------------------
// Causal flash attention per head. Qh/Kh/Vh pre-scaled f16 [S, D].
// grid = (S/64, H); block = 128 threads = 4 waves; each wave owns 16 q-rows.
// KV processed in blocks of 64 with online softmax.
// ---------------------------------------------------------------------------
__global__ __launch_bounds__(128)
void flash_attn(const _Float16* __restrict__ Qh, const _Float16* __restrict__ Kh,
                const _Float16* __restrict__ Vh, _Float16* __restrict__ Ah) {
    __shared__ __align__(16) _Float16 Vlds[128][64];   // [dk][kv] transposed
    __shared__ __align__(16) _Float16 Plds[4][16][64]; // per-wave P staging

    const int qb    = blockIdx.x * 64;
    const int h     = blockIdx.y;
    const int hBase = h * DK_DIM;
    const int tid   = threadIdx.x;
    const int lane  = tid & 31;
    const int wave  = tid >> 5;      // 0..3
    const int l15   = lane & 15;
    const int half  = lane >> 4;
    const int q0    = qb + wave * 16;

    // ---- preload Q fragments (A layout), DK=128 -> 4 k-slices ----
    V16H qf[4];
#pragma unroll
    for (int ks = 0; ks < 4; ++ks) {
        const _Float16* qp = &Qh[(size_t)(q0 + l15) * D_DIM + hBase + ks * 32 + half * 16];
        qf[ks].h[0] = *(const v8h*)qp;
        qf[ks].h[1] = *(const v8h*)(qp + 8);
    }

    v8f  O[8];
#pragma unroll
    for (int j = 0; j < 8; ++j) O[j] = (v8f){};
    float mrow[8], lrow[8];
#pragma unroll
    for (int v = 0; v < 8; ++v) { mrow[v] = -1e30f; lrow[v] = 0.f; }

    for (int kv = 0; kv < qb + 64; kv += 64) {
        // ---- stage V block transposed in LDS ----
        __syncthreads();
#pragma unroll 4
        for (int i = 0; i < 16; ++i) {
            int rowIn = i * 4 + wave;                 // 0..63
            int colB  = (tid & 31) * 4;               // 0..124
            const v4h vv = *(const v4h*)&Vh[(size_t)(kv + rowIn) * D_DIM + hBase + colB];
#pragma unroll
            for (int j = 0; j < 4; ++j) Vlds[colB + j][rowIn] = vv[j];
        }
        __syncthreads();

        if (kv + 64 < S_DIM)  // prefetch next K block rows (global_prefetch_b8)
            __builtin_prefetch(&Kh[(size_t)(kv + 64 + l15) * D_DIM + hBase], 0, 0);

        // ---- S = Q K^T : B fragments straight from global Kh ----
        v8f St[4];
#pragma unroll
        for (int t = 0; t < 4; ++t) St[t] = (v8f){};
#pragma unroll
        for (int t = 0; t < 4; ++t)
#pragma unroll
            for (int ks = 0; ks < 4; ++ks) {
                const v16h bf = *(const v16h*)
                    &Kh[(size_t)(kv + t * 16 + l15) * D_DIM + hBase + ks * 32 + half * 16];
                St[t] = wmma_f16(qf[ks].v, bf, St[t]);
            }

        // ---- causal mask (elementwise only when block overlaps diagonal) ----
        if (kv + 63 > q0) {
#pragma unroll
            for (int t = 0; t < 4; ++t) {
                const int n = kv + t * 16 + l15;
#pragma unroll
                for (int v = 0; v < 8; ++v) {
                    const int m = q0 + half * 8 + v;
                    if (n > m) St[t][v] = -1e30f;
                }
            }
        }

        // ---- online softmax: row max / exp / row sum ----
        float mnew[8], alpha[8], rs[8];
#pragma unroll
        for (int v = 0; v < 8; ++v) {
            float mx = St[0][v];
#pragma unroll
            for (int t = 1; t < 4; ++t) mx = fmaxf(mx, St[t][v]);
#pragma unroll
            for (int off = 8; off >= 1; off >>= 1)
                mx = fmaxf(mx, __shfl_xor(mx, off, 32));
            mnew[v]  = fmaxf(mrow[v], mx);
            alpha[v] = __expf(mrow[v] - mnew[v]);
            rs[v]    = 0.f;
        }
#pragma unroll
        for (int t = 0; t < 4; ++t)
#pragma unroll
            for (int v = 0; v < 8; ++v) {
                float p = __expf(St[t][v] - mnew[v]);
                St[t][v] = p;
                rs[v] += p;
            }
#pragma unroll
        for (int v = 0; v < 8; ++v) {
#pragma unroll
            for (int off = 8; off >= 1; off >>= 1)
                rs[v] += __shfl_xor(rs[v], off, 32);
            lrow[v] = lrow[v] * alpha[v] + rs[v];
            mrow[v] = mnew[v];
        }
#pragma unroll
        for (int j = 0; j < 8; ++j)
#pragma unroll
            for (int v = 0; v < 8; ++v) O[j][v] *= alpha[v];

        // ---- P -> per-wave LDS (C layout -> A layout round trip) ----
#pragma unroll
        for (int t = 0; t < 4; ++t)
#pragma unroll
            for (int v = 0; v < 8; ++v)
                Plds[wave][half * 8 + v][t * 16 + l15] = (_Float16)St[t][v];

        V16H pf[2];
#pragma unroll
        for (int ks = 0; ks < 2; ++ks) {
            const _Float16* p = &Plds[wave][l15][ks * 32 + half * 16];
            pf[ks].h[0] = *(const v8h*)p;
            pf[ks].h[1] = *(const v8h*)(p + 8);
        }

        // ---- O += P @ V ----
#pragma unroll
        for (int j = 0; j < 8; ++j)
#pragma unroll
            for (int ks = 0; ks < 2; ++ks) {
                const v16h bf = *(const v16h*)&Vlds[j * 16 + l15][ks * 32 + half * 16];
                O[j] = wmma_f16(pf[ks].v, bf, O[j]);
            }
    }

    // ---- normalize and store attn output (f16) ----
    float invl[8];
#pragma unroll
    for (int v = 0; v < 8; ++v) invl[v] = 1.0f / lrow[v];
#pragma unroll
    for (int j = 0; j < 8; ++j) {
        const int col = hBase + j * 16 + l15;
#pragma unroll
        for (int v = 0; v < 8; ++v)
            Ah[(size_t)(q0 + half * 8 + v) * D_DIM + col] = (_Float16)(O[j][v] * invl[v]);
    }
}

// ---------------------------------------------------------------------------
extern "C" void kernel_launch(void* const* d_in, const int* in_sizes, int n_in,
                              void* d_out, int out_size, void* d_ws, size_t ws_size,
                              hipStream_t stream) {
    (void)in_sizes; (void)n_in; (void)out_size; (void)ws_size;
    const float* query = (const float*)d_in[0];
    const float* key_  = (const float*)d_in[1];
    const float* value = (const float*)d_in[2];
    const float* Wq    = (const float*)d_in[3];
    const float* Wk    = (const float*)d_in[4];
    const float* Wv    = (const float*)d_in[5];
    const float* Wo    = (const float*)d_in[6];

    const size_t mat = (size_t)S_DIM * D_DIM;
    _Float16* Qh = (_Float16*)d_ws;
    _Float16* Kh = Qh + mat;
    _Float16* Vh = Kh + mat;
    _Float16* Ah = Vh + mat;

    const float inv = 0.08838834764831845f;  // 1/sqrt(128); applied to BOTH q and k
    const dim3 gg(D_DIM / 128, S_DIM / 128);

    gemm_xwT<false, true><<<gg, 256, 0, stream>>>(query, Wq, Qh, inv);
    gemm_xwT<false, true><<<gg, 256, 0, stream>>>(key_,  Wk, Kh, inv);
    gemm_xwT<false, true><<<gg, 256, 0, stream>>>(value, Wv, Vh, 1.0f);

    flash_attn<<<dim3(S_DIM / 64, H_NUM), 128, 0, stream>>>(Qh, Kh, Vh, Ah);

    gemm_xwT<true, false><<<gg, 256, 0, stream>>>(Ah, Wo, (float*)d_out, 1.0f);
}
